// Decoder_45655502356680
// MI455X (gfx1250) — matrix operand order repaired and checked by reading, not compiled, not run on hardware
//
#include <hip/hip_runtime.h>
#include <math.h>

// ---------------------------------------------------------------------------
// LSTM decoder (T=32, B=64, D=H=512, V=32000) for MI455X / gfx1250.
//
//   * All GEMMs lowered to v_wmma_f32_16x16x32_bf16 (bf16 in, fp32 accum).
//   * Vocab GEMM [2048 x 32000, K=512]: 128x128 block tiles, A/B panels
//     staged in 139 KB of dynamic LDS (gfx1250 WGP has 320 KB), 8 waves per
//     block each computing a 32x64 register tile -> each weight byte is read
//     from L2 once per M-block (16x reuse) instead of once per 16-row tile.
//   * Serial recurrence: per step, a 64x2048 (K=1024) WMMA GEMM (32x64 per
//     wave, register-blocked) + elementwise LSTM cell. hx kept bf16, cx fp32.
//   * Weights (33 MB bf16 vocab + 4 MB gates) are L2-resident (192 MB L2).
// ---------------------------------------------------------------------------

typedef __bf16 bf16;
typedef __attribute__((ext_vector_type(16))) __bf16 v16bf;
typedef __attribute__((ext_vector_type(8)))  __bf16 v8bf;
typedef __attribute__((ext_vector_type(8)))  float  v8f;

union V16U { v16bf v; v8bf h[2]; };

#define TT 32
#define BB 64
#define DD 512
#define HH 512
#define G4H 2048          // 4*H
#define VV 32000
#define SS 16

#define KC  256           // K-panel chunk staged in LDS (2 passes over K=512)
#define RSC 272           // padded LDS row stride (elems): 136 dwords -> +8 banks/row
#define VOC_LDS_BYTES (2 * 128 * RSC * 2)   // A panel + B panel = 139,264 B

// ----------------------------- setup kernels ------------------------------

__global__ void k_f32_to_bf16(const float* __restrict__ src,
                              bf16* __restrict__ dst, int n) {
  int i = blockIdx.x * blockDim.x + threadIdx.x;
  if (i < n) dst[i] = (bf16)src[i];
}

// Wcat[j, 0:512] = W_ih[j, :]; Wcat[j, 512:1024] = W_hh[j, :]  (bf16)
__global__ void k_build_wcat(const float* __restrict__ Wih,
                             const float* __restrict__ Whh,
                             bf16* __restrict__ Wcat) {
  int i = blockIdx.x * blockDim.x + threadIdx.x;  // [0, 2048*1024)
  if (i >= G4H * 1024) return;
  int j = i >> 10, k = i & 1023;
  float v = (k < DD) ? Wih[j * DD + k] : Whh[j * HH + (k - DD)];
  Wcat[i] = (bf16)v;
}

__global__ void k_build_bias(const float* __restrict__ bih,
                             const float* __restrict__ bhh,
                             float* __restrict__ bias) {
  int i = blockIdx.x * blockDim.x + threadIdx.x;
  if (i < G4H) bias[i] = bih[i] + bhh[i];
}

// hx0 = encoder_outputs[S-1]  (to bf16), cx0 = 0
__global__ void k_init_state(const float* __restrict__ enc,
                             bf16* __restrict__ hxbf,
                             float* __restrict__ cx) {
  int i = blockIdx.x * blockDim.x + threadIdx.x;
  if (i < BB * HH) {
    hxbf[i] = (bf16)enc[(size_t)(SS - 1) * BB * HH + i];
    cx[i] = 0.0f;
  }
}

// --------------------------- recurrence kernels ---------------------------
// Gates GEMM: Gf[64, 2048] = [x_t | hx] (bf16, K=1024) @ Wcat^T + bias
// 64 waves (8 blocks x 8): mt2 in [0,2) x nt in [0,32); 32x64 tile per wave.
__global__ void k_gates_gemm(const bf16* __restrict__ Xbf,
                             const bf16* __restrict__ hxbf,
                             const bf16* __restrict__ Wcat,
                             const float* __restrict__ bias,
                             float* __restrict__ Gf, int t) {
  const int lane = threadIdx.x & 31;
  const int wave = blockIdx.x * (blockDim.x >> 5) + (threadIdx.x >> 5);
  const int nt  = wave & 31;     // 32 tiles of 64 cols -> 2048
  const int mt2 = wave >> 5;     // 2 tiles of 32 rows  -> 64
  const int half = lane >> 4;
  const int lm = lane & 15;

  const bf16* xrow[2];
  const bf16* hrow[2];
#pragma unroll
  for (int i = 0; i < 2; ++i) {
    const int m = mt2 * 32 + i * 16 + lm;              // batch row
    xrow[i] = Xbf + ((size_t)t * BB + m) * DD;
    hrow[i] = hxbf + (size_t)m * HH;
  }

  v8f acc[2][4] = {};
  for (int kt = 0; kt < 32; ++kt) {                    // K = 1024
    const int k0 = kt * 32;
    V16U a[2];
#pragma unroll
    for (int i = 0; i < 2; ++i) {
      const bf16* arow = (k0 < DD) ? (xrow[i] + k0) : (hrow[i] + (k0 - DD));
      a[i].h[0] = *(const v8bf*)(arow + 16 * half);
      a[i].h[1] = *(const v8bf*)(arow + 16 * half + 8);
    }
#pragma unroll
    for (int s = 0; s < 4; ++s) {
      const int n = nt * 64 + s * 16 + lm;             // gate column
      const v16bf b = *(const v16bf*)(Wcat + (size_t)n * 1024 + k0 + 16 * half);
      acc[0][s] = __builtin_amdgcn_wmma_f32_16x16x32_bf16(
          false, a[0].v, false, b, (short)0, acc[0][s], false, false);
      acc[1][s] = __builtin_amdgcn_wmma_f32_16x16x32_bf16(
          false, a[1].v, false, b, (short)0, acc[1][s], false, false);
    }
  }
#pragma unroll
  for (int i = 0; i < 2; ++i) {
#pragma unroll
    for (int s = 0; s < 4; ++s) {
      const int n = nt * 64 + s * 16 + lm;
      const float bv = bias[n];
#pragma unroll
      for (int r = 0; r < 8; ++r) {
        const int mm = mt2 * 32 + i * 16 + half * 8 + r;
        Gf[(size_t)mm * G4H + n] = acc[i][s][r] + bv;
      }
    }
  }
}

// Elementwise LSTM cell over [B, H]; gates in PyTorch order (i, f, g, o).
__global__ void k_lstm_cell(const float* __restrict__ Gf,
                            float* __restrict__ cx,
                            bf16* __restrict__ hxbf,
                            bf16* __restrict__ hseq, int t) {
  int idx = blockIdx.x * blockDim.x + threadIdx.x;  // [0, 64*512)
  if (idx >= BB * HH) return;
  int b = idx >> 9, h = idx & (HH - 1);
  const float* g = Gf + (size_t)b * G4H;
  float gi = g[h];
  float gf = g[HH + h];
  float gg = g[2 * HH + h];
  float go = g[3 * HH + h];
  float si = 1.0f / (1.0f + __expf(-gi));
  float sf = 1.0f / (1.0f + __expf(-gf));
  float so = 1.0f / (1.0f + __expf(-go));
  float c = sf * cx[idx] + si * tanhf(gg);
  cx[idx] = c;
  float hn = so * tanhf(c);
  bf16 hb = (bf16)hn;
  hxbf[idx] = hb;
  hseq[(size_t)t * (BB * HH) + idx] = hb;            // row m = t*64 + b
}

// ----------------------------- vocab kernels ------------------------------
// logits[b, t, v] = hseq[m=t*64+b, :] @ Wvoc[v, :]^T + b_voc[v]
// 4000 blocks (16 mtiles x 250 ntiles), 128x128 per block, 32x64 per wave.
// A/B K-panels (KC=256 wide) staged in dynamic LDS, two passes over K=512.
__global__ void k_vocab_gemm(const bf16* __restrict__ hseq,
                             const bf16* __restrict__ Wvoc,
                             const float* __restrict__ bvoc,
                             float* __restrict__ out) {
  extern __shared__ char smem[];
  bf16* As = (bf16*)smem;              // 128 rows x RSC (KC cols + pad)
  bf16* Bs = As + 128 * RSC;           // 128 rows x RSC

  const int mt = blockIdx.x / 250;     // 16 tiles of 128 rows -> 2048
  const int nt = blockIdx.x % 250;     // 250 tiles of 128 cols -> 32000
  const int m0 = mt * 128;
  const int n0 = nt * 128;
  const int tid = threadIdx.x;

  const int lane = tid & 31;
  const int w  = tid >> 5;             // wave 0..7
  const int wm = w >> 1;               // 4 x 32-row slices
  const int wn = w & 1;                // 2 x 64-col slices
  const int half = lane >> 4;
  const int lm = lane & 15;

  const bf16* a0 = As + (wm * 32 + lm) * RSC;       // m-subtile 0 LDS row
  const bf16* a1 = a0 + 16 * RSC;                   // m-subtile 1 LDS row
  const bf16* br = Bs + (wn * 64 + lm) * RSC;       // + s*16*RSC per subtile

  v8f acc[2][4] = {};
  for (int kc = 0; kc < 2; ++kc) {                  // K panels: [0,256),[256,512)
    const int kbase = kc * KC;
    // Cooperative fill: 128 rows x 256 cols per panel; each chunk = 8 bf16.
    // 32 consecutive threads read 512 contiguous bytes of one global row.
    for (int c = tid; c < 128 * (KC / 8); c += 256) {
      const int row = c >> 5;                        // KC/8 = 32 chunks per row
      const int col = (c & 31) << 3;
      *(v8bf*)(As + row * RSC + col) =
          *(const v8bf*)(hseq + (size_t)(m0 + row) * HH + kbase + col);
      *(v8bf*)(Bs + row * RSC + col) =
          *(const v8bf*)(Wvoc + (size_t)(n0 + row) * HH + kbase + col);
    }
    __syncthreads();

#pragma unroll
    for (int kt = 0; kt < KC / 32; ++kt) {
      const int k0 = kt * 32;
      V16U a[2];
      a[0].h[0] = *(const v8bf*)(a0 + k0 + 16 * half);
      a[0].h[1] = *(const v8bf*)(a0 + k0 + 16 * half + 8);
      a[1].h[0] = *(const v8bf*)(a1 + k0 + 16 * half);
      a[1].h[1] = *(const v8bf*)(a1 + k0 + 16 * half + 8);
#pragma unroll
      for (int s = 0; s < 4; ++s) {
        const v16bf b = *(const v16bf*)(br + s * 16 * RSC + k0 + 16 * half);
        acc[0][s] = __builtin_amdgcn_wmma_f32_16x16x32_bf16(
            false, a[0].v, false, b, (short)0, acc[0][s], false, false);
        acc[1][s] = __builtin_amdgcn_wmma_f32_16x16x32_bf16(
            false, a[1].v, false, b, (short)0, acc[1][s], false, false);
      }
    }
    __syncthreads();
  }

#pragma unroll
  for (int i = 0; i < 2; ++i) {
#pragma unroll
    for (int s = 0; s < 4; ++s) {
      const int n = n0 + wn * 64 + s * 16 + lm;
      const float bv = bvoc[n];
#pragma unroll
      for (int r = 0; r < 8; ++r) {
        const int m = m0 + wm * 32 + i * 16 + half * 8 + r;   // m = t*64 + b
        const int tt2 = m >> 6;
        const int bb2 = m & 63;
        out[((size_t)bb2 * TT + tt2) * VV + n] = acc[i][s][r] + bv;
      }
    }
  }
}

// One block per output row (b*T + t); argmax over V = 32000.
__global__ void k_argmax(const float* __restrict__ logits,
                         float* __restrict__ pred) {
  __shared__ float smax[256];
  __shared__ int   sidx[256];
  const int row = blockIdx.x;
  const float* p = logits + (size_t)row * VV;
  float best = -INFINITY;
  int   bi = 0;
  for (int v = threadIdx.x; v < VV; v += blockDim.x) {
    float x = p[v];
    if (x > best || (x == best && v < bi)) { best = x; bi = v; }
  }
  smax[threadIdx.x] = best;
  sidx[threadIdx.x] = bi;
  __syncthreads();
  for (int off = 128; off > 0; off >>= 1) {
    if (threadIdx.x < off) {
      float o = smax[threadIdx.x + off];
      int   oi = sidx[threadIdx.x + off];
      if (o > smax[threadIdx.x] ||
          (o == smax[threadIdx.x] && oi < sidx[threadIdx.x])) {
        smax[threadIdx.x] = o;
        sidx[threadIdx.x] = oi;
      }
    }
    __syncthreads();
  }
  if (threadIdx.x == 0) pred[row] = (float)sidx[0];
}

// ------------------------------- launcher ---------------------------------

extern "C" void kernel_launch(void* const* d_in, const int* in_sizes, int n_in,
                              void* d_out, int out_size, void* d_ws,
                              size_t ws_size, hipStream_t stream) {
  (void)in_sizes; (void)n_in; (void)out_size; (void)ws_size;
  const float* X    = (const float*)d_in[0];   // [T,B,D]
  const float* enc  = (const float*)d_in[1];   // [S,B,H]
  const float* Wih  = (const float*)d_in[2];   // [4H,D]
  const float* Whh  = (const float*)d_in[3];   // [4H,H]
  const float* bih  = (const float*)d_in[4];   // [4H]
  const float* bhh  = (const float*)d_in[5];   // [4H]
  const float* Wvoc = (const float*)d_in[6];   // [V,H]
  const float* bvoc = (const float*)d_in[7];   // [V]
  float* out = (float*)d_out;                  // logits [B,T,V] ++ predicts [B,T]

  // Workspace carve-out (~42 MB), 256-byte aligned chunks.
  char* ws = (char*)d_ws;
  size_t off = 0;
  auto carve = [&](size_t bytes) -> void* {
    void* p = ws + off;
    off = (off + bytes + 255) & ~(size_t)255;
    return p;
  };
  bf16*  Xbf    = (bf16*)carve((size_t)TT * BB * DD * 2);     // 2 MB
  bf16*  WcatBf = (bf16*)carve((size_t)G4H * 1024 * 2);       // 4 MB
  bf16*  WvocBf = (bf16*)carve((size_t)VV * HH * 2);          // 32.8 MB
  float* bias   = (float*)carve((size_t)G4H * 4);             // 8 KB
  float* Gf     = (float*)carve((size_t)BB * G4H * 4);        // 512 KB
  float* cx     = (float*)carve((size_t)BB * HH * 4);         // 128 KB
  bf16*  hxbf   = (bf16*)carve((size_t)BB * HH * 2);          // 64 KB
  bf16*  hseq   = (bf16*)carve((size_t)TT * BB * HH * 2);     // 2 MB

  const int nX  = TT * BB * DD;                 // 1,048,576
  const int nWv = VV * HH;                      // 16,384,000
  const int nWc = G4H * 1024;                   // 2,097,152

  k_f32_to_bf16<<<(nX + 255) / 256, 256, 0, stream>>>(X, Xbf, nX);
  k_f32_to_bf16<<<(nWv + 255) / 256, 256, 0, stream>>>(Wvoc, WvocBf, nWv);
  k_build_wcat<<<(nWc + 255) / 256, 256, 0, stream>>>(Wih, Whh, WcatBf);
  k_build_bias<<<(G4H + 255) / 256, 256, 0, stream>>>(bih, bhh, bias);
  k_init_state<<<(BB * HH + 255) / 256, 256, 0, stream>>>(enc, hxbf, cx);

  // Serial recurrence: 32 dependent steps.
  for (int t = 0; t < TT; ++t) {
    // 64 waves = 8 blocks x 8 waves: 2 mtiles(32) x 32 ntiles(64)
    k_gates_gemm<<<8, 256, 0, stream>>>(Xbf, hxbf, WcatBf, bias, Gf, t);
    k_lstm_cell<<<(BB * HH + 255) / 256, 256, 0, stream>>>(Gf, cx, hxbf, hseq, t);
  }

  // 4000 blocks x 8 waves: 16 mtiles(128) x 250 ntiles(128), LDS-staged K.
  k_vocab_gemm<<<4000, 256, VOC_LDS_BYTES, stream>>>(hseq, WvocBf, bvoc, out);

  float* pred = out + (size_t)BB * TT * VV;
  k_argmax<<<BB * TT, 256, 0, stream>>>(out, pred);
}
